// ProbSparseSelfAttention_37056977830135
// MI455X (gfx1250) — matrix operand order repaired and checked by reading, not compiled, not run on hardware
//
#include <hip/hip_runtime.h>
#include <hip/hip_bf16.h>

typedef __attribute__((ext_vector_type(16))) _Float16 v16h;
typedef __attribute__((ext_vector_type(8)))  _Float16 v8h;
typedef __attribute__((ext_vector_type(8)))  float    v8f;
typedef __attribute__((ext_vector_type(4)))  float    v4f;
typedef __attribute__((ext_vector_type(4)))  int      v4i;

#define D_MODEL 1024
#define L_SEQ   2048
#define N_HEADS 16
#define D_K     64
#define NROWS   4096   // B * L

// ---------------------------------------------------------------------------
// Fragment loaders (wave32 CDNA5 WMMA layouts, ISA 7.12.2)
// A 16x32 f16: lane m = lane&15, K = 16*hi + e (16 contiguous halves; this code's order)
// B 32x16 f16: lane n = lane&15, K = 16*hi + e  (16 contiguous halves; this code's order, same as A, not the ISA table)
// C 16x16 f32: vgpr r -> row r + 8*hi, col = lane&15
// ---------------------------------------------------------------------------

__device__ __forceinline__ v16h load_af16(const _Float16* p) {
  v8h lo = *(const v8h*)(p);
  v8h hi = *(const v8h*)(p + 8);
  v16h r;
#pragma unroll
  for (int i = 0; i < 8; ++i) { r[i] = lo[i]; r[i + 8] = hi[i]; }
  return r;
}

__device__ __forceinline__ v16h load_bf16(const _Float16* p) {
  v8h lo = *(const v8h*)(p);
  v8h hi = *(const v8h*)(p + 8);
  v16h r;
#pragma unroll
  for (int i = 0; i < 8; ++i) { r[i] = lo[i]; r[i + 8] = hi[i]; }
  return r;
}

__device__ __forceinline__ v16h cvt_af32(const float* p) {
  v4f c0 = *(const v4f*)(p);
  v4f c1 = *(const v4f*)(p + 4);
  v4f c2 = *(const v4f*)(p + 8);
  v4f c3 = *(const v4f*)(p + 12);
  v16h r;
#pragma unroll
  for (int i = 0; i < 4; ++i) {
    r[i]      = (_Float16)c0[i];
    r[i + 4]  = (_Float16)c1[i];
    r[i + 8]  = (_Float16)c2[i];
    r[i + 12] = (_Float16)c3[i];
  }
  return r;
}

__device__ __forceinline__ v8f wmma16(v16h a, v16h b, v8f c) {
  return __builtin_amdgcn_wmma_f32_16x16x32_f16(false, a, false, b,
                                                (short)0, c, false, false);
}

// ---------------------------------------------------------------------------
// Async global->LDS staging (CDNA5 GLOBAL_LOAD_ASYNC_TO_LDS_B128, ASYNCcnt),
// with a synchronous fallback if the builtin is unavailable.
// ---------------------------------------------------------------------------
#if __has_builtin(__builtin_amdgcn_global_load_async_to_lds_b128)
#define ASYNC_STAGE 1
typedef __attribute__((address_space(1))) v4i* as1_v4i;
typedef __attribute__((address_space(3))) v4i* as3_v4i;
#else
#define ASYNC_STAGE 0
#endif

__device__ __forceinline__ void stage_b128(const _Float16* gsrc, _Float16* ldst) {
#if ASYNC_STAGE
  __builtin_amdgcn_global_load_async_to_lds_b128((as1_v4i)gsrc, (as3_v4i)ldst,
                                                 0, 0);
#else
  *(v8h*)ldst = *(const v8h*)gsrc;
#endif
}

__device__ __forceinline__ void stage_wait() {
#if ASYNC_STAGE
#if __has_builtin(__builtin_amdgcn_s_wait_asynccnt)
  __builtin_amdgcn_s_wait_asynccnt(0);
#else
  asm volatile("s_wait_asynccnt 0" ::: "memory");
#endif
#endif
}

// ---------------------------------------------------------------------------
// Kernel 1: weight transpose + f32->f16   Wt[n][k] = (f16) W[k][n]
// ---------------------------------------------------------------------------
__global__ __launch_bounds__(256) void wtrans_kernel(
    const float* __restrict__ Wq, const float* __restrict__ Wk,
    const float* __restrict__ Wv, const float* __restrict__ Wo,
    _Float16* __restrict__ wqt, _Float16* __restrict__ wkt,
    _Float16* __restrict__ wvt, _Float16* __restrict__ wot) {
  __shared__ float t[16][17];
  const float* W;
  _Float16* Wt;
  switch (blockIdx.z) {
    case 0:  W = Wq; Wt = wqt; break;
    case 1:  W = Wk; Wt = wkt; break;
    case 2:  W = Wv; Wt = wvt; break;
    default: W = Wo; Wt = wot; break;
  }
  const int kb = blockIdx.y * 16, nb = blockIdx.x * 16;
  const int tx = threadIdx.x, ty = threadIdx.y;
  t[ty][tx] = W[(size_t)(kb + ty) * D_MODEL + nb + tx];
  __syncthreads();
  Wt[(size_t)(nb + ty) * D_MODEL + kb + tx] = (_Float16)t[tx][ty];
}

// ---------------------------------------------------------------------------
// Kernel 2: projection GEMM  Y = X(f32) @ W + b,  scattered into head layout
//   vmode==0 : dst[b][h][lq][d]   (Q, K)
//   vmode==1 : dst[b][h][d][lq]   (V, transposed for P@V B-fragments)
// Block = 8 waves x 16 rows, shared 64-col range; B-tile (64x32 f16, 4KB)
// double-buffered through LDS with async loads.
// ---------------------------------------------------------------------------
__global__ __launch_bounds__(256) void proj_kernel(
    const float* __restrict__ X, const _Float16* __restrict__ Wt,
    const float* __restrict__ bias, _Float16* __restrict__ dst, int vmode) {
  __shared__ __align__(16) _Float16 sB[2][64][32];

  const int tid  = threadIdx.x;
  const int lane = tid & 31;
  const int wv   = tid >> 5;
  const int hi   = lane >> 4;
  const int ln   = lane & 15;
  const int m0   = blockIdx.y * 128 + wv * 16;
  const int n0   = blockIdx.x * 64;

  const int srow = tid >> 2;          // 0..63 (one 16B chunk per thread)
  const int soff = (tid & 3) * 8;     // halves

  const float* Xr = X + (size_t)(m0 + ln) * D_MODEL;
  v8f acc[4] = {v8f{}, v8f{}, v8f{}, v8f{}};

  // prologue: stage k=0 tile into buffer 0
  stage_b128(Wt + (size_t)(n0 + srow) * D_MODEL + soff, &sB[0][srow][soff]);
  stage_wait();
  __syncthreads();

  for (int k = 0; k < D_MODEL; k += 32) {
    const int cur = (k >> 5) & 1;
    if (k + 32 < D_MODEL)
      stage_b128(Wt + (size_t)(n0 + srow) * D_MODEL + (k + 32) + soff,
                 &sB[cur ^ 1][srow][soff]);

    v16h a = cvt_af32(Xr + k + 16 * hi);
#pragma unroll
    for (int j = 0; j < 4; ++j) {
      v16h b = load_bf16(&sB[cur][16 * j + ln][16 * hi]);
      acc[j] = wmma16(a, b, acc[j]);
    }
    stage_wait();
    __syncthreads();
  }

#pragma unroll
  for (int j = 0; j < 4; ++j) {
    const int c = n0 + 16 * j + ln;
    const float bval = bias[c];
    const int h = c >> 6, d = c & 63;
#pragma unroll
    for (int r = 0; r < 8; ++r) {
      const int gi = m0 + r + 8 * hi;       // global row (b*L + lq)
      const int bb = gi >> 11, lq = gi & (L_SEQ - 1);
      const _Float16 val = (_Float16)(acc[j][r] + bval);
      if (vmode == 0)
        dst[(((size_t)(bb * N_HEADS + h)) * L_SEQ + lq) * D_K + d] = val;
      else
        dst[(((size_t)(bb * N_HEADS + h)) * D_K + d) * L_SEQ + lq] = val;
    }
  }
}

// ---------------------------------------------------------------------------
// Kernel 3: flash attention per (b,h). 4 waves x 16 query rows per block.
// K (32x64) and V (64x32) chunk tiles are shared by all 4 waves ->
// double-buffered async staging through LDS. Per-wave P tile relayout in LDS.
// ---------------------------------------------------------------------------
__global__ __launch_bounds__(128) void attn_kernel(
    const _Float16* __restrict__ Qh, const _Float16* __restrict__ Kh,
    const _Float16* __restrict__ Vt, _Float16* __restrict__ Oc) {
  __shared__ __align__(16) _Float16 sK[2][32][64];   // [key][d]
  __shared__ __align__(16) _Float16 sV[2][64][32];   // [d][key]
  __shared__ __align__(16) _Float16 sP[4][16][32];

  const int tid  = threadIdx.x;
  const int lane = tid & 31;
  const int wv   = tid >> 5;
  const int hi   = lane >> 4;
  const int ln   = lane & 15;
  const int bh   = blockIdx.y;          // b*16 + h
  const int b    = bh >> 4;
  const int h    = bh & 15;
  const int q0   = blockIdx.x * 64 + wv * 16;

  const _Float16* Qp = Qh + ((size_t)bh * L_SEQ + q0) * D_K;
  const _Float16* Kp = Kh + (size_t)bh * L_SEQ * D_K;
  const _Float16* Vp = Vt + (size_t)bh * D_K * L_SEQ;

  const v16h qa0 = load_af16(Qp + ln * D_K + 16 * hi);
  const v16h qa1 = load_af16(Qp + ln * D_K + 32 + 16 * hi);

  v8f o0 = {}, o1 = {}, o2 = {}, o3 = {};
  float mrow[8], lrow[8];
#pragma unroll
  for (int r = 0; r < 8; ++r) { mrow[r] = -3.0e38f; lrow[r] = 0.f; }

  // stage K/V chunk [kc..kc+31] into buffer `buf` (128 threads, 2 chunks each)
  auto stageKV = [&](int kc, int buf) {
#pragma unroll
    for (int c = tid; c < 256; c += 128) {
      const int krow = c >> 3, koff = (c & 7) * 8;   // 32 rows x 8 chunks
      stage_b128(Kp + (size_t)(kc + krow) * D_K + koff, &sK[buf][krow][koff]);
    }
#pragma unroll
    for (int c = tid; c < 256; c += 128) {
      const int vrow = c >> 2, voff = (c & 3) * 8;   // 64 rows x 4 chunks
      stage_b128(Vp + (size_t)vrow * L_SEQ + kc + voff, &sV[buf][vrow][voff]);
    }
  };

  stageKV(0, 0);
  stage_wait();
  __syncthreads();

  for (int kc = 0; kc < L_SEQ; kc += 32) {
    const int cur = (kc >> 5) & 1;
    if (kc + 32 < L_SEQ) stageKV(kc + 32, cur ^ 1);

    // ---- S tiles (16 queries x 32 keys), K-dim = 64 -> 2 wmma each
    v8f s0 = {}, s1 = {};
    {
      v16h kb;
      kb = load_bf16(&sK[cur][ln][16 * hi]);
      s0 = wmma16(qa0, kb, s0);
      kb = load_bf16(&sK[cur][ln][32 + 16 * hi]);
      s0 = wmma16(qa1, kb, s0);
      kb = load_bf16(&sK[cur][16 + ln][16 * hi]);
      s1 = wmma16(qa0, kb, s1);
      kb = load_bf16(&sK[cur][16 + ln][32 + 16 * hi]);
      s1 = wmma16(qa1, kb, s1);
    }
    // ---- online softmax (butterflies stay within each 16-lane half)
    float p0[8], p1[8], alpha[8];
#pragma unroll
    for (int r = 0; r < 8; ++r) {
      const float a  = s0[r] * 0.125f;      // 1/sqrt(64)
      const float bb = s1[r] * 0.125f;
      float rm = fmaxf(a, bb);
      rm = fmaxf(rm, __shfl_xor(rm, 1, 32));
      rm = fmaxf(rm, __shfl_xor(rm, 2, 32));
      rm = fmaxf(rm, __shfl_xor(rm, 4, 32));
      rm = fmaxf(rm, __shfl_xor(rm, 8, 32));
      const float mnew = fmaxf(mrow[r], rm);
      alpha[r] = __expf(mrow[r] - mnew);
      p0[r]    = __expf(a - mnew);
      p1[r]    = __expf(bb - mnew);
      float rs = p0[r] + p1[r];
      rs += __shfl_xor(rs, 1, 32);
      rs += __shfl_xor(rs, 2, 32);
      rs += __shfl_xor(rs, 4, 32);
      rs += __shfl_xor(rs, 8, 32);
      lrow[r] = lrow[r] * alpha[r] + rs;
      mrow[r] = mnew;
    }
    // ---- rescale O, spill P (f16) to per-wave LDS tile [16][32]
#pragma unroll
    for (int r = 0; r < 8; ++r) {
      o0[r] *= alpha[r]; o1[r] *= alpha[r];
      o2[r] *= alpha[r]; o3[r] *= alpha[r];
      sP[wv][r + 8 * hi][ln]      = (_Float16)p0[r];
      sP[wv][r + 8 * hi][ln + 16] = (_Float16)p1[r];
    }
    asm volatile("s_wait_dscnt 0" ::: "memory");   // same-wave LDS RAW
    // ---- reload P as A fragment, accumulate O += P @ V from staged tile
    const v16h pa = load_af16(&sP[wv][ln][16 * hi]);
    v16h vb;
    vb = load_bf16(&sV[cur][ln][16 * hi]);
    o0 = wmma16(pa, vb, o0);
    vb = load_bf16(&sV[cur][16 + ln][16 * hi]);
    o1 = wmma16(pa, vb, o1);
    vb = load_bf16(&sV[cur][32 + ln][16 * hi]);
    o2 = wmma16(pa, vb, o2);
    vb = load_bf16(&sV[cur][48 + ln][16 * hi]);
    o3 = wmma16(pa, vb, o3);

    stage_wait();
    __syncthreads();
  }

  // ---- normalize and store to Oc[b][lq][h*64 + d] (f16)
  _Float16* Op = Oc + (size_t)b * L_SEQ * D_MODEL;
#pragma unroll
  for (int r = 0; r < 8; ++r) {
    const float inv = 1.0f / lrow[r];
    const int row = q0 + r + 8 * hi;
    const size_t base = (size_t)row * D_MODEL + h * D_K + ln;
    Op[base]      = (_Float16)(o0[r] * inv);
    Op[base + 16] = (_Float16)(o1[r] * inv);
    Op[base + 32] = (_Float16)(o2[r] * inv);
    Op[base + 48] = (_Float16)(o3[r] * inv);
  }
}

// ---------------------------------------------------------------------------
// Kernel 4: output projection  out(f32) = Oc(f16) @ Wo + bo
// Same LDS-staged structure as proj_kernel.
// ---------------------------------------------------------------------------
__global__ __launch_bounds__(256) void out_gemm_kernel(
    const _Float16* __restrict__ Oc, const _Float16* __restrict__ Wot,
    const float* __restrict__ bias, float* __restrict__ out) {
  __shared__ __align__(16) _Float16 sB[2][64][32];

  const int tid  = threadIdx.x;
  const int lane = tid & 31;
  const int wv   = tid >> 5;
  const int hi   = lane >> 4;
  const int ln   = lane & 15;
  const int m0   = blockIdx.y * 128 + wv * 16;
  const int n0   = blockIdx.x * 64;

  const int srow = tid >> 2;
  const int soff = (tid & 3) * 8;

  const _Float16* Or = Oc + (size_t)(m0 + ln) * D_MODEL;
  v8f acc[4] = {v8f{}, v8f{}, v8f{}, v8f{}};

  stage_b128(Wot + (size_t)(n0 + srow) * D_MODEL + soff, &sB[0][srow][soff]);
  stage_wait();
  __syncthreads();

  for (int k = 0; k < D_MODEL; k += 32) {
    const int cur = (k >> 5) & 1;
    if (k + 32 < D_MODEL)
      stage_b128(Wot + (size_t)(n0 + srow) * D_MODEL + (k + 32) + soff,
                 &sB[cur ^ 1][srow][soff]);

    v16h a = load_af16(Or + k + 16 * hi);
#pragma unroll
    for (int j = 0; j < 4; ++j) {
      v16h b = load_bf16(&sB[cur][16 * j + ln][16 * hi]);
      acc[j] = wmma16(a, b, acc[j]);
    }
    stage_wait();
    __syncthreads();
  }

#pragma unroll
  for (int j = 0; j < 4; ++j) {
    const int c = n0 + 16 * j + ln;
    const float bval = bias[c];
#pragma unroll
    for (int r = 0; r < 8; ++r) {
      const int gi = m0 + r + 8 * hi;
      out[(size_t)gi * D_MODEL + c] = acc[j][r] + bval;
    }
  }
}

// ---------------------------------------------------------------------------
extern "C" void kernel_launch(void* const* d_in, const int* in_sizes, int n_in,
                              void* d_out, int out_size, void* d_ws, size_t ws_size,
                              hipStream_t stream) {
  (void)in_sizes; (void)n_in; (void)out_size; (void)ws_size;
  const float* q  = (const float*)d_in[0];
  const float* k  = (const float*)d_in[1];
  const float* v  = (const float*)d_in[2];
  const float* Wq = (const float*)d_in[3];
  const float* bq = (const float*)d_in[4];
  const float* Wk = (const float*)d_in[5];
  const float* bk = (const float*)d_in[6];
  const float* Wv = (const float*)d_in[7];
  const float* bv = (const float*)d_in[8];
  const float* Wo = (const float*)d_in[9];
  const float* bo = (const float*)d_in[10];

  // workspace layout (f16 elements): 4x 1M weights, 3x 4M heads, 1x 4M Oc
  _Float16* ws  = (_Float16*)d_ws;
  const size_t M1 = (size_t)1 << 20;
  _Float16* wqt = ws;
  _Float16* wkt = ws + 1 * M1;
  _Float16* wvt = ws + 2 * M1;
  _Float16* wot = ws + 3 * M1;
  _Float16* Qh  = ws + 4 * M1;
  _Float16* Kh  = ws + 8 * M1;
  _Float16* Vt  = ws + 12 * M1;
  _Float16* Oc  = ws + 16 * M1;

  // 1) transpose + convert the four weight matrices
  wtrans_kernel<<<dim3(64, 64, 4), dim3(16, 16), 0, stream>>>(
      Wq, Wk, Wv, Wo, wqt, wkt, wvt, wot);

  // 2) Q/K/V projections (wave tile 16x64; grid covers 4096 x 1024)
  dim3 pg(D_MODEL / 64, NROWS / 128);
  proj_kernel<<<pg, 256, 0, stream>>>(q, wqt, bq, Qh, 0);
  proj_kernel<<<pg, 256, 0, stream>>>(k, wkt, bk, Kh, 0);
  proj_kernel<<<pg, 256, 0, stream>>>(v, wvt, bv, Vt, 1);

  // 3) flash attention: 4 waves/block * 16 rows = 64 queries per block
  attn_kernel<<<dim3(L_SEQ / 64, 2 * N_HEADS), 128, 0, stream>>>(Qh, Kh, Vt, Oc);

  // 4) output projection -> f32 d_out
  out_gemm_kernel<<<pg, 256, 0, stream>>>(Oc, wot, bo, (float*)d_out);
}
